// DMNEncoder_61967788147169
// MI455X (gfx1250) — matrix operand order repaired and checked by reading, not compiled, not run on hardware
//
#include <hip/hip_runtime.h>
#include <hip/hip_bf16.h>

// ---------------------------------------------------------------------------
// DMN encoder for MI455X (gfx1250, wave32).
//  - All GEMMs via V_WMMA_F32_16X16X32_BF16 (bf16 A/B, f32 accum).
//  - Weights repacked once into per-lane WMMA B fragments.
//  - Sequential GRU scans keep h/gh AND the whole Whh fragment set in LDS;
//    Whh is staged with GLOBAL_LOAD_ASYNC_TO_LDS_B128 when available.
// ---------------------------------------------------------------------------

typedef __attribute__((ext_vector_type(16))) __bf16 bf16x16;
typedef __attribute__((ext_vector_type(8)))  __bf16 bf16x8;
typedef __attribute__((ext_vector_type(8)))  float  f32x8;
typedef __attribute__((ext_vector_type(4)))  int    v4i_;

#define NTILES 24   // 384 / 16
#define GDIM   384
#define WFRAG_ELEMS (4 * NTILES * 512)   // Whh fragments: 4 kt x 24 nt x 512

__device__ __forceinline__ float sigmoidf_(float x) {
  return 1.0f / (1.0f + __expf(-x));
}

__device__ __forceinline__ f32x8 wmma_bf16(bf16x16 a, bf16x16 b, f32x8 c) {
  // (neg_a, A, neg_b, B, c_mod, C, reuse_a, reuse_b)
  return __builtin_amdgcn_wmma_f32_16x16x32_bf16(false, a, false, b,
                                                 (short)0, c, false, false);
}

// B fragment: 16 contiguous bf16 per lane (packed by repack_w).
__device__ __forceinline__ bf16x16 load_bfrag(const __bf16* p) {
  union { bf16x16 v; bf16x8 h[2]; } u;
  u.h[0] = *(const bf16x8*)(p);
  u.h[1] = *(const bf16x8*)(p + 8);
  return u.v;
}

// A fragment (16x32 bf16) from row-major f32 (stride ld), converting to bf16.
// ISA layout: row = lane&15, k(e) = (e&7) + (e>>3)*16 + (lane>>4)*8.
__device__ __forceinline__ bf16x16 load_afrag_f32(const float* base, int ld,
                                                  int kt, int lane) {
  int row = lane & 15;
  int kb  = kt * 32 + ((lane >> 4) << 3);
  const float* p = base + row * ld + kb;
  bf16x16 a;
#pragma unroll
  for (int i = 0; i < 8; ++i) {
    a[i]     = (__bf16)p[i];
    a[i + 8] = (__bf16)p[i + 16];
  }
  return a;
}

// A fragment from row-major bf16 LDS (stride ld elements).
__device__ __forceinline__ bf16x16 load_afrag_bf16(const __bf16* base, int ld,
                                                   int kt, int lane) {
  int row = lane & 15;
  int kb  = kt * 32 + ((lane >> 4) << 3);
  const __bf16* p = base + row * ld + kb;
  union { bf16x16 v; bf16x8 h[2]; } u;
  u.h[0] = *(const bf16x8*)(p);
  u.h[1] = *(const bf16x8*)(p + 16);
  return u.v;
}

// ---------------------------------------------------------------------------
// Stage `chunks` x 16B from global into LDS. Uses the CDNA5 async-to-LDS
// path (ASYNCcnt + s_wait_asynccnt) when the builtin exists; otherwise plain
// vector copies. Ends with a workgroup barrier either way.
// ---------------------------------------------------------------------------
__device__ __forceinline__ void stage_to_lds(const __bf16* __restrict__ g,
                                             __bf16* l, int chunks, int tid,
                                             int nthreads) {
#if __has_builtin(__builtin_amdgcn_global_load_async_to_lds_b128)
  for (int i = tid; i < chunks; i += nthreads) {
    auto* gp = (__attribute__((address_space(1))) v4i_*)
        ((const char*)g + (size_t)i * 16);
    auto* lp = (__attribute__((address_space(3))) v4i_*)
        ((char*)l + (size_t)i * 16);
    __builtin_amdgcn_global_load_async_to_lds_b128(gp, lp, 0, 0);
  }
#if __has_builtin(__builtin_amdgcn_s_wait_asynccnt)
  __builtin_amdgcn_s_wait_asynccnt(0);
#else
  asm volatile("s_wait_asynccnt 0" ::: "memory");
#endif
#else
  const int4* gs = (const int4*)g;
  int4* ls = (int4*)l;
  for (int i = tid; i < chunks; i += nthreads) ls[i] = gs[i];
#endif
  __syncthreads();
}

// ---------------------------------------------------------------------------
// Repack weight W (384 x ldW f32, row-major) into WMMA B fragments:
// out[(kt*24+nt)*512 + lane*16 + e] with n = lane&15,
// k = kt*32 + (e&7) + ((e>>3)<<4) + ((lane>>4)<<3)  (B 32x16 bf16 layout). Zero-pads k >= ldW.
// grid = KT*24 blocks, 512 threads.
// ---------------------------------------------------------------------------
__global__ void repack_w(const float* __restrict__ W, __bf16* __restrict__ out,
                         int ldW) {
  int frag = blockIdx.x;            // kt*NTILES + nt
  int kt = frag / NTILES;
  int nt = frag - kt * NTILES;
  int tid  = threadIdx.x;           // 0..511
  int lane = tid >> 4;
  int e    = tid & 15;
  int n = lane & 15;
  int k = kt * 32 + (e & 7) + ((e >> 3) << 4) + ((lane >> 4) << 3);
  int g = nt * 16 + n;
  float v = (k < ldW) ? W[(size_t)g * ldW + k] : 0.0f;
  out[(size_t)frag * 512 + lane * 16 + e] = (__bf16)v;
}

// ---------------------------------------------------------------------------
// Row GEMM: outC[row, 0:384] = A[row, 0:K] @ Bfrag + bias.
// If tokens != nullptr: A row = Asrc + tokens[row]*K (embedding gather).
// Block = 256 threads handles 64 rows. LDS-staged bf16 A (K padded to LDK).
// ---------------------------------------------------------------------------
__global__ __launch_bounds__(256) void gemm_rows(
    const float* __restrict__ Asrc, const int* __restrict__ tokens,
    int K, int LDK, int KT,
    const __bf16* __restrict__ Bfrag, const float* __restrict__ bias,
    float* __restrict__ outC) {
  __shared__ alignas(16) __bf16 sA[64 * 320];
  int tid = threadIdx.x;
  int rowBase = blockIdx.x * 64;

  int total = 64 * LDK;
  for (int i = tid; i < total; i += 256) {
    int r = i / LDK;
    int c = i - r * LDK;
    float v = 0.0f;
    if (c < K) {
      const float* src = tokens
          ? (Asrc + (size_t)tokens[rowBase + r] * K)
          : (Asrc + (size_t)(rowBase + r) * K);
      v = src[c];
    }
    sA[r * LDK + c] = (__bf16)v;
  }
  __syncthreads();

  int wave = tid >> 5;
  int lane = tid & 31;
  int msub = wave & 3;               // 4 M sub-tiles of 16 rows
  int nt0  = (wave >> 2) * 12;       // 2 wave-halves x 12 N tiles
  const __bf16* aBase = &sA[msub * 16 * LDK];

  for (int j = 0; j < 12; ++j) {
    int nt = nt0 + j;
    f32x8 acc = {};
    const __bf16* bBase = Bfrag + (size_t)nt * 512 + lane * 16;
    for (int kt = 0; kt < KT; ++kt) {
      __builtin_prefetch(bBase + (size_t)(kt + 1) * NTILES * 512, 0, 1);
      bf16x16 a = load_afrag_bf16(aBase, LDK, kt, lane);
      bf16x16 b = load_bfrag(bBase + (size_t)kt * NTILES * 512);
      acc = wmma_bf16(a, b, acc);
    }
    // D layout: vgpr i -> m = i + (lane>=16 ? 8 : 0), n = lane&15
    int n  = lane & 15;
    int mb = (lane >> 4) << 3;
    int col = nt * 16 + n;
    float bv = bias[col];
    int row0 = rowBase + msub * 16 + mb;
#pragma unroll
    for (int i = 0; i < 8; ++i)
      outC[(size_t)(row0 + i) * GDIM + col] = acc[i] + bv;
  }
}

// ---------------------------------------------------------------------------
// gh = h @ Whh^T for 16 batch rows resident in LDS (f32 [16][128]).
// 8 waves x 3 N tiles = 24 tiles; K = 128 (4 k-steps). Whh fragments read
// from LDS (sW). Results -> gh LDS.
// ---------------------------------------------------------------------------
__device__ __forceinline__ void gh_gemm(const float* hLds, float* ghLds,
                                        const __bf16* sW, int tid) {
  int wave = tid >> 5;
  int lane = tid & 31;
  int nt0 = wave * 3;
  f32x8 acc0 = {}, acc1 = {}, acc2 = {};
#pragma unroll
  for (int kt = 0; kt < 4; ++kt) {
    bf16x16 a = load_afrag_f32(hLds, 128, kt, lane);
    const __bf16* bb = sW + (size_t)(kt * NTILES) * 512 + lane * 16;
    acc0 = wmma_bf16(a, load_bfrag(bb + (size_t)(nt0 + 0) * 512), acc0);
    acc1 = wmma_bf16(a, load_bfrag(bb + (size_t)(nt0 + 1) * 512), acc1);
    acc2 = wmma_bf16(a, load_bfrag(bb + (size_t)(nt0 + 2) * 512), acc2);
  }
  int n  = lane & 15;
  int mb = (lane >> 4) << 3;
#pragma unroll
  for (int i = 0; i < 8; ++i) {
    ghLds[(mb + i) * GDIM + (nt0 + 0) * 16 + n] = acc0[i];
    ghLds[(mb + i) * GDIM + (nt0 + 1) * 16 + n] = acc1[i];
    ghLds[(mb + i) * GDIM + (nt0 + 2) * 16 + n] = acc2[i];
  }
}

// ---------------------------------------------------------------------------
// Sequential GRU scan over T steps for 16 batch rows per block (4 blocks).
// gi already contains x@Wih^T + b_ih, laid out [b*T + t][384].
// Whh fragments staged to LDS once (async-to-LDS when available).
// sentsOut (if set): emit h at t%64==63 -> sents[b][t/64][128].
// finalOut/finalOut2 (if set): final h (q_vecs).
// ---------------------------------------------------------------------------
__global__ __launch_bounds__(256) void gru_scan(
    const float* __restrict__ gi, const __bf16* __restrict__ WhhFrag,
    const float* __restrict__ bhh, int T,
    float* __restrict__ sentsOut, float* __restrict__ finalOut,
    float* __restrict__ finalOut2) {
  __shared__ alignas(16) __bf16 sW[WFRAG_ELEMS];   // 96 KB
  __shared__ float h[16 * 128];
  __shared__ float gh[16 * GDIM];
  int tid = threadIdx.x;
  int b0 = blockIdx.x * 16;

  for (int i = tid; i < 2048; i += 256) h[i] = 0.0f;
  stage_to_lds(WhhFrag, sW, WFRAG_ELEMS * 2 / 16, tid, 256);  // barriers

  for (int t = 0; t < T; ++t) {
    gh_gemm(h, gh, sW, tid);
    __syncthreads();

    int e0 = tid * 8;
    int bl = e0 >> 7;
    int j0 = e0 & 127;
    const float* giRow = gi + ((size_t)(b0 + bl) * T + t) * GDIM;
    float* hRow = h + bl * 128;
    const float* ghRow = gh + bl * GDIM;
    bool emit = (sentsOut != nullptr) && ((t & 63) == 63);
    float* sRow =
        emit ? (sentsOut + ((size_t)(b0 + bl) * 16 + (t >> 6)) * 128) : nullptr;
#pragma unroll
    for (int jj = 0; jj < 8; ++jj) {
      int j = j0 + jj;
      float r  = sigmoidf_(giRow[j]       + ghRow[j]       + bhh[j]);
      float z  = sigmoidf_(giRow[128 + j] + ghRow[128 + j] + bhh[128 + j]);
      float nn = tanhf(giRow[256 + j] + r * (ghRow[256 + j] + bhh[256 + j]));
      float hn = (1.0f - z) * nn + z * hRow[j];
      hRow[j] = hn;
      if (emit) sRow[j] = hn;
    }
    __syncthreads();
  }

  if (finalOut) {
    for (int i = tid; i < 2048; i += 256) {
      finalOut[(size_t)b0 * 128 + i] = h[i];
      if (finalOut2) finalOut2[(size_t)b0 * 128 + i] = h[i];
    }
  }
}

// ---------------------------------------------------------------------------
// Episodic memory: 3 iterations of gates + 16-step gated GRU.
// gi_ep = sents @ ep_Wih^T + b_ih precomputed (mem-invariant).
// 4 blocks x 16 batch rows; mem/h/gh/Whh resident in LDS.
// ---------------------------------------------------------------------------
__global__ __launch_bounds__(256) void episode(
    const float* __restrict__ sents, const float* __restrict__ qvec,
    const float* __restrict__ giEp, const __bf16* __restrict__ WhhFrag,
    const float* __restrict__ bhh, const float* __restrict__ gate_w,
    const float* __restrict__ gate_b,
    float* __restrict__ epOut, float* __restrict__ gatesOut) {
  __shared__ alignas(16) __bf16 sW[WFRAG_ELEMS];   // 96 KB
  __shared__ float mem[16 * 128];
  __shared__ float h[16 * 128];
  __shared__ float gh[16 * GDIM];
  __shared__ float gts[256];
  int tid = threadIdx.x;
  int b0 = blockIdx.x * 16;

  for (int i = tid; i < 2048; i += 256)
    mem[i] = qvec[(size_t)b0 * 128 + i];
  stage_to_lds(WhhFrag, sW, WFRAG_ELEMS * 2 / 16, tid, 256);  // barriers

  for (int it = 0; it < 3; ++it) {
    // gates[b][s] = sigmoid(feats . gate_w + gate_b); one thread per (b,s).
    {
      int bl = tid >> 4;
      int s  = tid & 15;
      const float* sr = sents + ((size_t)(b0 + bl) * 16 + s) * 128;
      const float* qr = qvec + (size_t)(b0 + bl) * 128;
      const float* mr = mem + bl * 128;
      float acc = gate_b[0];
      for (int j = 0; j < 128; ++j) {
        float sv = sr[j], qv = qr[j], mv = mr[j];
        float d1 = sv - qv, d2 = sv - mv;
        acc += sv * qv * gate_w[j] + sv * mv * gate_w[128 + j] +
               d1 * d1 * gate_w[256 + j] + d2 * d2 * gate_w[384 + j];
      }
      gts[tid] = sigmoidf_(acc);
    }
    for (int i = tid; i < 2048; i += 256) h[i] = 0.0f;
    __syncthreads();

    for (int s = 0; s < 16; ++s) {
      gh_gemm(h, gh, sW, tid);
      __syncthreads();

      int e0 = tid * 8;
      int bl = e0 >> 7;
      int j0 = e0 & 127;
      const float* giRow = giEp + ((size_t)(b0 + bl) * 16 + s) * GDIM;
      float* hRow = h + bl * 128;
      const float* ghRow = gh + bl * GDIM;
      float w = gts[bl * 16 + s];
#pragma unroll
      for (int jj = 0; jj < 8; ++jj) {
        int j = j0 + jj;
        float r  = sigmoidf_(giRow[j]       + ghRow[j]       + bhh[j]);
        float z  = sigmoidf_(giRow[128 + j] + ghRow[128 + j] + bhh[128 + j]);
        float nn = tanhf(giRow[256 + j] + r * (ghRow[256 + j] + bhh[256 + j]));
        float hnew = (1.0f - z) * nn + z * hRow[j];
        hRow[j] = w * hnew + (1.0f - w) * hRow[j];
      }
      __syncthreads();
    }
    for (int i = tid; i < 2048; i += 256) mem[i] = h[i];
    __syncthreads();
  }

  for (int i = tid; i < 2048; i += 256)
    epOut[(size_t)b0 * 128 + i] = h[i];
  {
    int bl = tid >> 4;
    int s  = tid & 15;
    gatesOut[(size_t)(b0 + bl) * 16 + s] = gts[tid];
  }
}

// ---------------------------------------------------------------------------
// Host orchestration.
// Inputs (setup_inputs order):
//  0 x_stories(i32 B*L) 1 x_story_masks(unused) 2 x_questions(i32 B*QL)
//  3 n_sents(unused) 4 emb 5 enc_w_ih 6 enc_w_hh 7 enc_b_ih 8 enc_b_hh
//  9 gate_w 10 gate_b 11 ep_w_ih 12 ep_w_hh 13 ep_b_ih 14 ep_b_hh
// Output: q_vecs(8192) | ep_state(8192) | gates(1024)  (f32)
// ---------------------------------------------------------------------------
extern "C" void kernel_launch(void* const* d_in, const int* in_sizes, int n_in,
                              void* d_out, int out_size, void* d_ws,
                              size_t ws_size, hipStream_t stream) {
  const int*   x_stories   = (const int*)d_in[0];
  const int*   x_questions = (const int*)d_in[2];
  const float* emb      = (const float*)d_in[4];
  const float* enc_w_ih = (const float*)d_in[5];
  const float* enc_w_hh = (const float*)d_in[6];
  const float* enc_b_ih = (const float*)d_in[7];
  const float* enc_b_hh = (const float*)d_in[8];
  const float* gate_w   = (const float*)d_in[9];
  const float* gate_b   = (const float*)d_in[10];
  const float* ep_w_ih  = (const float*)d_in[11];
  const float* ep_w_hh  = (const float*)d_in[12];
  const float* ep_b_ih  = (const float*)d_in[13];
  const float* ep_b_hh  = (const float*)d_in[14];
  (void)in_sizes; (void)n_in; (void)out_size; (void)ws_size;

  char* ws = (char*)d_ws;
  size_t off = 0;
  auto carve = [&](size_t bytes) -> void* {
    off = (off + 255) & ~(size_t)255;
    void* p = ws + off;
    off += bytes;
    return p;
  };
  float*  gi_s  = (float*)carve((size_t)65536 * 384 * 4);   // story gi
  float*  gi_q  = (float*)carve((size_t)2048 * 384 * 4);    // question gi
  float*  gi_ep = (float*)carve((size_t)1024 * 384 * 4);    // episode gi
  __bf16* wihF  = (__bf16*)carve((size_t)10 * 24 * 512 * 2);
  __bf16* whhF  = (__bf16*)carve((size_t)4 * 24 * 512 * 2);
  __bf16* ewihF = (__bf16*)carve((size_t)4 * 24 * 512 * 2);
  __bf16* ewhhF = (__bf16*)carve((size_t)4 * 24 * 512 * 2);
  float*  sents = (float*)carve((size_t)64 * 16 * 128 * 4);
  float*  qvecw = (float*)carve((size_t)64 * 128 * 4);

  float* out = (float*)d_out;

  // Weight repack -> WMMA B fragments (bf16).
  repack_w<<<10 * 24, 512, 0, stream>>>(enc_w_ih, wihF, 300);
  repack_w<<<4 * 24, 512, 0, stream>>>(enc_w_hh, whhF, 128);
  repack_w<<<4 * 24, 512, 0, stream>>>(ep_w_ih, ewihF, 128);
  repack_w<<<4 * 24, 512, 0, stream>>>(ep_w_hh, ewhhF, 128);

  // Input projections (embedding gather fused into GEMM).
  gemm_rows<<<1024, 256, 0, stream>>>(emb, x_stories, 300, 320, 10, wihF,
                                      enc_b_ih, gi_s);
  gemm_rows<<<32, 256, 0, stream>>>(emb, x_questions, 300, 320, 10, wihF,
                                    enc_b_ih, gi_q);

  // Recurrent scans (batch split across 4 blocks each).
  gru_scan<<<4, 256, 0, stream>>>(gi_s, whhF, enc_b_hh, 1024, sents, nullptr,
                                  nullptr);
  gru_scan<<<4, 256, 0, stream>>>(gi_q, whhF, enc_b_hh, 32, nullptr, qvecw,
                                  out /* q_vecs */);

  // Episode input projection (mem-invariant).
  gemm_rows<<<16, 256, 0, stream>>>(sents, nullptr, 128, 128, 4, ewihF,
                                    ep_b_ih, gi_ep);

  // Episodic memory iterations.
  episode<<<4, 256, 0, stream>>>(sents, qvecw, gi_ep, ewhhF, ep_b_hh, gate_w,
                                 gate_b, out + 8192, out + 16384);
}
